// GNNLayer_7043746365793
// MI455X (gfx1250) — matrix operand order repaired and checked by reading, not compiled, not run on hardware
//
#include <hip/hip_runtime.h>
#include <math.h>

typedef __attribute__((ext_vector_type(16))) _Float16 v16h;
typedef __attribute__((ext_vector_type(8)))  float    v8f;

#define MIN_NORM_F 1e-15f
#define BALL_EPS_F 0.004f
#define MIN_C_F    1e-6f

static __device__ __forceinline__ float tanh_clip(float x) {
  return tanhf(fminf(fmaxf(x, -15.f), 15.f));
}

// A-fragment (16x32 f16): lane half kg holds K kg*16..kg*16+15 (contiguous)
// of row m = lane&15.  kc selects K-chunk (0 -> K 0..31, 1 -> K 32..63).
static __device__ __forceinline__ v16h load_a_f32(const float* __restrict__ row,
                                                  int kc, int kg) {
  const float* p0 = row + kc * 32 + kg * 16;
  const float* p1 = p0 + 8;
  v16h a;
#pragma unroll
  for (int i = 0; i < 8; ++i) {
    a[i]     = (_Float16)p0[i];
    a[i + 8] = (_Float16)p1[i];
  }
  return a;
}

// B-fragment (32x16 f16) for x@W^T: column N = nt*16 + n is weight row N;
// lane half kg holds K (kc*32 + kg*16) .. +15, contiguous in the LDS copy.
static __device__ __forceinline__ v16h load_b_lds(const _Float16* __restrict__ W,
                                                  int nt, int n, int kg, int kc) {
  const _Float16* p = W + (nt * 16 + n) * 64 + kc * 32 + kg * 16;
  v16h b;
#pragma unroll
  for (int i = 0; i < 16; ++i) b[i] = p[i];
  return b;
}

static __device__ __forceinline__ v8f wmma_f16(v16h a, v16h b, v8f c) {
  return __builtin_amdgcn_wmma_f32_16x16x32_f16(false, a, false, b,
                                                (short)0, c, false, false);
}

__global__ void zero2_kernel(float* __restrict__ a, float* __restrict__ b, long n) {
  long i = (long)blockIdx.x * blockDim.x + threadIdx.x;
  long stride = (long)gridDim.x * blockDim.x;
  for (; i < n; i += stride) { a[i] = 0.f; b[i] = 0.f; }
}

__global__ __launch_bounds__(256) void edge_kernel(
    const float* __restrict__ hidden, const float* __restrict__ path_state,
    const int* __restrict__ q_rel, const int* __restrict__ sub_idx,
    const int* __restrict__ rel_idx, const int* __restrict__ obj_idx,
    const int* __restrict__ r_idx, const float* __restrict__ rela,
    const float* __restrict__ Ws, const float* __restrict__ Wr,
    const float* __restrict__ Wqr, const float* __restrict__ Wqrb,
    const float* __restrict__ wa, const float* __restrict__ wab,
    const float* __restrict__ Wpp, const float* __restrict__ Wpr,
    const float* __restrict__ curv,
    float* __restrict__ msg_agg, float* __restrict__ path_out, int E) {
  // f16 copies of the five 64x64 weight matrices: 40 KB LDS
  __shared__ _Float16 sW[5 * 64 * 64];
  {
    const float* mats[5] = {Ws, Wr, Wqr, Wpp, Wpr};
    for (int idx = threadIdx.x; idx < 5 * 4096; idx += blockDim.x)
      sW[idx] = (_Float16)mats[idx >> 12][idx & 4095];
  }
  __syncthreads();

  const int lane = threadIdx.x & 31;
  const int wave = threadIdx.x >> 5;
  const long tile = (long)blockIdx.x * (blockDim.x >> 5) + wave;
  const long base = tile * 16;
  if (base >= E) return;

  const int m  = lane & 15;   // A row / D column within tile
  const int kg = lane >> 4;   // lane half

  // Gather pointers for A row m (both lane halves read the same row)
  int eA = (int)base + m;
  int idx0 = (eA < E) ? eA : 0;
  int subA = sub_idx[idx0];
  int relA = rel_idx[idx0];
  int qrA  = q_rel[r_idx[idx0]];
  const float* hsRow = hidden     + (long)subA * 64;
  const float* hpRow = path_state + (long)subA * 64;
  const float* hrRow = rela       + (long)relA * 64;
  const float* hqRow = rela       + (long)qrA  * 64;
  __builtin_prefetch(hsRow, 0, 0);
  __builtin_prefetch(hpRow, 0, 0);

  v8f accA[4], accP[4];
#pragma unroll
  for (int nt = 0; nt < 4; ++nt) {
    float bn = Wqrb[nt * 16 + m];  // bias broadcast over rows
#pragma unroll
    for (int r = 0; r < 8; ++r) { accA[nt][r] = bn; accP[nt][r] = 0.f; }
  }

#pragma unroll
  for (int kc = 0; kc < 2; ++kc) {
    v16h a_hs = load_a_f32(hsRow, kc, kg);
    v16h a_hr = load_a_f32(hrRow, kc, kg);
    v16h a_hq = load_a_f32(hqRow, kc, kg);
    v16h a_hp = load_a_f32(hpRow, kc, kg);
#pragma unroll
    for (int nt = 0; nt < 4; ++nt) {
      accA[nt] = wmma_f16(a_hs, load_b_lds(sW + 0 * 4096, nt, m, kg, kc), accA[nt]);
      accA[nt] = wmma_f16(a_hr, load_b_lds(sW + 1 * 4096, nt, m, kg, kc), accA[nt]);
      accA[nt] = wmma_f16(a_hq, load_b_lds(sW + 2 * 4096, nt, m, kg, kc), accA[nt]);
      accP[nt] = wmma_f16(a_hp, load_b_lds(sW + 3 * 4096, nt, m, kg, kc), accP[nt]);
      accP[nt] = wmma_f16(a_hr, load_b_lds(sW + 4 * 4096, nt, m, kg, kc), accP[nt]);
    }
  }

  // alpha = sigmoid(relu(attn) . wa + b); reduce over the 64 N values:
  // per D register r, lanes 0-15 (resp 16-31) hold row r (resp r+8).
  float wa4[4];
#pragma unroll
  for (int nt = 0; nt < 4; ++nt) wa4[nt] = wa[nt * 16 + m];
  const float bAlpha = wab[0];
  float alphaV[8];
#pragma unroll
  for (int r = 0; r < 8; ++r) {
    float s = 0.f;
#pragma unroll
    for (int nt = 0; nt < 4; ++nt) s += fmaxf(accA[nt][r], 0.f) * wa4[nt];
    s += __shfl_xor(s, 1, 32);
    s += __shfl_xor(s, 2, 32);
    s += __shfl_xor(s, 4, 32);
    s += __shfl_xor(s, 8, 32);
    alphaV[r] = 1.f / (1.f + __expf(-(s + bAlpha)));
  }

  // obj for this lane's D rows (row = r + 8*kg)
  int objArr[8];
#pragma unroll
  for (int r = 0; r < 8; ++r) {
    int e = (int)base + r + 8 * kg;
    objArr[r] = (e < E) ? obj_idx[e] : -1;
  }

  // scatter alpha * tanh(path_edge)
#pragma unroll
  for (int r = 0; r < 8; ++r) {
    int o = objArr[r];
    if (o < 0) continue;
    float al = alphaV[r];
#pragma unroll
    for (int nt = 0; nt < 4; ++nt)
      atomicAdd(path_out + (long)o * 64 + nt * 16 + m, al * tanhf(accP[nt][r]));
  }

  // hyperbolic message per edge: each lane owns dims {lane, lane+32}
  const float c    = fmaxf(curv[0], MIN_C_F);
  const float sc   = sqrtf(c);
  const float maxn = (1.f - BALL_EPS_F) / sc;
#pragma unroll
  for (int e = 0; e < 16; ++e) {
    int eg = (int)base + e;
    if (eg >= E) continue;
    float al = __shfl(alphaV[e & 7], (e & 8) ? 16 : 0, 32);
    int sub = sub_idx[eg];
    int rel = rel_idx[eg];
    int obj = obj_idx[eg];
    const float* xs = hidden + (long)sub * 64;
    const float* ys = rela   + (long)rel * 64;
    float x0 = xs[lane], x1 = xs[lane + 32];
    float y0 = ys[lane], y1 = ys[lane + 32];
    float px = x0 * x0 + x1 * x1;
    float py = y0 * y0 + y1 * y1;
    float pxy = x0 * y0 + x1 * y1;
#pragma unroll
    for (int msk = 1; msk < 32; msk <<= 1) {
      px  += __shfl_xor(px,  msk, 32);
      py  += __shfl_xor(py,  msk, 32);
      pxy += __shfl_xor(pxy, msk, 32);
    }
    float xn = fmaxf(sqrtf(fmaxf(px, 0.f)), MIN_NORM_F);
    float yn = fmaxf(sqrtf(fmaxf(py, 0.f)), MIN_NORM_F);
    float sx = tanh_clip(sc * xn) / (sc * xn);   // expmap0 scale of hs
    float sy = tanh_clip(sc * yn) / (sc * yn);   // expmap0 scale of hr
    float vx = fmaxf(fabsf(sx) * xn, MIN_NORM_F);
    if (vx > maxn) sx *= maxn / vx;              // project
    float vy = fmaxf(fabsf(sy) * yn, MIN_NORM_F);
    if (vy > maxn) sy *= maxn / vy;
    float X2 = sx * sx * px, Y2 = sy * sy * py, XY = sx * sy * pxy;
    float Aa = 1.f + 2.f * c * XY + c * Y2;
    float Bb = 1.f - c * X2;
    float den = fmaxf(1.f + 2.f * c * XY + c * c * X2 * Y2, MIN_NORM_F);
    float ca = Aa * sx / den, cb = Bb * sy / den;  // mobius: ca*hs + cb*hr
    float mn2 = ca * ca * px + 2.f * ca * cb * pxy + cb * cb * py;
    float mn = fmaxf(sqrtf(fmaxf(mn2, 0.f)), MIN_NORM_F);
    float g = (mn > maxn) ? (maxn / mn) : 1.f;     // project
    float mnp = fmaxf(fminf(mn, maxn), MIN_NORM_F);
    float arg = fminf(fmaxf(sc * mnp, -1.f + 1e-5f), 1.f - 1e-5f);
    float fs = g * atanhf(arg) / (sc * mnp);       // logmap0 scale
    float m0 = fs * (ca * x0 + cb * y0);
    float m1 = fs * (ca * x1 + cb * y1);
    atomicAdd(msg_agg + (long)obj * 64 + lane,      al * m0);
    atomicAdd(msg_agg + (long)obj * 64 + lane + 32, al * m1);
  }
}

__global__ __launch_bounds__(256) void node_kernel(
    const float* __restrict__ msg_agg, const float* __restrict__ Wh,
    const float* __restrict__ curv, float* __restrict__ out_hidden, int n_node) {
  __shared__ _Float16 sW[64 * 64];
  for (int idx = threadIdx.x; idx < 4096; idx += blockDim.x)
    sW[idx] = (_Float16)Wh[idx];
  __syncthreads();

  const int lane = threadIdx.x & 31;
  const int wave = threadIdx.x >> 5;
  const long tile = (long)blockIdx.x * (blockDim.x >> 5) + wave;
  const long base = tile * 16;
  if (base >= n_node) return;

  const int m  = lane & 15;
  const int kg = lane >> 4;
  long row = base + m;
  if (row >= n_node) row = n_node - 1;
  const float* arow = msg_agg + row * 64;

  v8f acc[4];
#pragma unroll
  for (int nt = 0; nt < 4; ++nt)
#pragma unroll
    for (int r = 0; r < 8; ++r) acc[nt][r] = 0.f;

#pragma unroll
  for (int kc = 0; kc < 2; ++kc) {
    v16h a = load_a_f32(arow, kc, kg);
#pragma unroll
    for (int nt = 0; nt < 4; ++nt)
      acc[nt] = wmma_f16(a, load_b_lds(sW, nt, m, kg, kc), acc[nt]);
  }

  const float c    = fmaxf(curv[0], MIN_C_F);
  const float sc   = sqrtf(c);
  const float maxn = (1.f - BALL_EPS_F) / sc;
#pragma unroll
  for (int r = 0; r < 8; ++r) {
    float s = 0.f;
#pragma unroll
    for (int nt = 0; nt < 4; ++nt) s += acc[nt][r] * acc[nt][r];
    s += __shfl_xor(s, 1, 32);
    s += __shfl_xor(s, 2, 32);
    s += __shfl_xor(s, 4, 32);
    s += __shfl_xor(s, 8, 32);
    float un = fmaxf(sqrtf(fmaxf(s, 0.f)), MIN_NORM_F);
    float s1 = tanh_clip(sc * un) / (sc * un);         // expmap0
    float vn = fmaxf(fabsf(s1) * un, MIN_NORM_F);
    float g = (vn > maxn) ? (maxn / vn) : 1.f;         // project
    float an = fmaxf(fminf(vn, maxn), MIN_NORM_F);
    float arg = fminf(fmaxf(sc * an, -1.f + 1e-5f), 1.f - 1e-5f);
    float scale = s1 * g * atanhf(arg) / (sc * an);    // logmap0
    int node = (int)base + r + 8 * kg;
    if (node < n_node) {
#pragma unroll
      for (int nt = 0; nt < 4; ++nt)
        out_hidden[(long)node * 64 + nt * 16 + m] = scale * acc[nt][r];
    }
  }
}

extern "C" void kernel_launch(void* const* d_in, const int* in_sizes, int n_in,
                              void* d_out, int out_size, void* d_ws, size_t ws_size,
                              hipStream_t stream) {
  const float* hidden     = (const float*)d_in[0];
  const float* path_state = (const float*)d_in[1];
  const int*   q_rel      = (const int*)d_in[2];
  const int*   sub_idx    = (const int*)d_in[3];
  const int*   rel_idx    = (const int*)d_in[4];
  const int*   obj_idx    = (const int*)d_in[5];
  const int*   r_idx      = (const int*)d_in[6];
  const float* rela       = (const float*)d_in[7];
  const float* Ws         = (const float*)d_in[8];
  const float* Wr         = (const float*)d_in[9];
  const float* Wqr        = (const float*)d_in[10];
  const float* Wqrb       = (const float*)d_in[11];
  const float* wa         = (const float*)d_in[12];
  const float* wab        = (const float*)d_in[13];
  const float* Wh         = (const float*)d_in[14];
  const float* Wpp        = (const float*)d_in[15];
  const float* Wpr        = (const float*)d_in[16];
  const float* curv       = (const float*)d_in[17];

  const int E      = in_sizes[3];
  const int n_node = in_sizes[0] / 64;

  float* msg_agg    = (float*)d_ws;                       // [n_node,64] f32 scratch
  float* out_hidden = (float*)d_out;                      // [n_node,64]
  float* path_out   = (float*)d_out + (size_t)n_node * 64;// [n_node,64]

  long nz = (long)n_node * 64;
  int zg = (int)((nz + 255) / 256);
  if (zg > 4096) zg = 4096;
  zero2_kernel<<<zg, 256, 0, stream>>>(msg_agg, path_out, nz);

  const int wavesPerBlock = 8;  // 256 threads
  long etiles = ((long)E + 15) / 16;
  int egrid = (int)((etiles + wavesPerBlock - 1) / wavesPerBlock);
  edge_kernel<<<egrid, 256, 0, stream>>>(hidden, path_state, q_rel, sub_idx,
      rel_idx, obj_idx, r_idx, rela, Ws, Wr, Wqr, Wqrb, wa, wab, Wpp, Wpr,
      curv, msg_agg, path_out, E);

  long ntiles = ((long)n_node + 15) / 16;
  int ngrid = (int)((ntiles + wavesPerBlock - 1) / wavesPerBlock);
  node_kernel<<<ngrid, 256, 0, stream>>>(msg_agg, Wh, curv, out_hidden, n_node);
}
